// LTLOperator_16492674417183
// MI455X (gfx1250) — matrix operand order repaired and checked by reading, not compiled, not run on hardware
//
#include <hip/hip_runtime.h>

// Problem constants (match reference)
constexpr int kB = 64;
constexpr int kT = 4096;
constexpr int kV = 128;
constexpr int kF = 128;
constexpr float kSlope = 1.0f;
constexpr float kAlpha = 0.2f;
constexpr float kLog2e = 1.4426950408889634f;

typedef __attribute__((ext_vector_type(16))) __bf16 v16bf;
typedef __attribute__((ext_vector_type(8)))  float  v8f;

// Build a 16x32 bf16 A-fragment for v_wmma_f32_16x16x32_bf16 from one f32 row.
// ISA layout (wave32): lanes 0-15 -> M=lane, K = {0..7, 16..23}+kt*32
//                      lanes 16-31 -> M=lane-16, K = {8..15, 24..31}+kt*32
__device__ inline v16bf cvt_a_frag(const float* __restrict__ rowp, int kt, int lane) {
  const int k0 = kt * 32 + ((lane & 16) ? 8 : 0);
  const float4* p = (const float4*)(rowp + k0);
  float4 x0 = p[0];
  float4 x1 = p[1];
  const float4* q = (const float4*)(rowp + k0 + 16);
  float4 y0 = q[0];
  float4 y1 = q[1];
  v16bf a;
  a[0]  = (__bf16)x0.x; a[1]  = (__bf16)x0.y; a[2]  = (__bf16)x0.z; a[3]  = (__bf16)x0.w;
  a[4]  = (__bf16)x1.x; a[5]  = (__bf16)x1.y; a[6]  = (__bf16)x1.z; a[7]  = (__bf16)x1.w;
  a[8]  = (__bf16)y0.x; a[9]  = (__bf16)y0.y; a[10] = (__bf16)y0.z; a[11] = (__bf16)y0.w;
  a[12] = (__bf16)y1.x; a[13] = (__bf16)y1.y; a[14] = (__bf16)y1.z; a[15] = (__bf16)y1.w;
  return a;
}

// Kernel 1: combined[b,t,f] = traces[b,t,:]@w_prop + shifted_traces[b,t,:]@w_metric + bias[f]
// Written to d_out as staging (scan kernel rewrites in place).
// One wave = one 16-row bt strip x all 128 F columns. 8 waves / block -> 128 rows / block.
__global__ __launch_bounds__(256)
void ltl_gemm_kernel(const float* __restrict__ traces,
                     const float* __restrict__ w_prop,
                     const float* __restrict__ w_metric,
                     const float* __restrict__ init_metric,
                     const float* __restrict__ bias,
                     float* __restrict__ out) {
  // Weights staged in LDS, bf16, fragment-linear layout:
  // sX[((nt*4 + kt)*32 + lane)*16 + j]  where  k = kt*32 + (lane>=16 ? 8:0) + (j&7) + (j>=8 ? 16:0),
  //                                            n = nt*16 + (lane & 15)
  __shared__ alignas(32) __bf16 sP[kV * kF];
  __shared__ alignas(32) __bf16 sM[kV * kF];

  const int tid = threadIdx.x;
  for (int idx = tid; idx < kV * kF; idx += 256) {
    const int j    = idx & 15;
    const int lane = (idx >> 4) & 31;
    const int frag = idx >> 9;        // 0..31
    const int kt   = frag & 3;
    const int nt   = frag >> 2;
    const int k = kt * 32 + ((lane & 16) ? 8 : 0) + (j & 7) + ((j & 8) ? 16 : 0);
    const int n = nt * 16 + (lane & 15);
    sP[idx] = (__bf16)w_prop[k * kF + n];
    sM[idx] = (__bf16)w_metric[k * kF + n];
  }
  __syncthreads();

  const int lane    = tid & 31;
  const int wave    = tid >> 5;
  const int rowbase = blockIdx.x * 128 + wave * 16;   // flattened b*t row
  const int r       = rowbase + (lane & 15);          // this lane's A row

  // Unshifted row (for w_prop) and t+1 row (for w_metric; batch tail -> init_metric)
  const float* ap  = traces + (size_t)r * kV;
  const float* apn = ((r % kT) == (kT - 1)) ? init_metric
                                            : traces + (size_t)(r + 1) * kV;

  v16bf a0[4], a1[4];
#pragma unroll
  for (int kt = 0; kt < 4; ++kt) {
    a0[kt] = cvt_a_frag(ap,  kt, lane);
    a1[kt] = cvt_a_frag(apn, kt, lane);
  }

  const int nlo   = lane & 15;
  const int mbase = (lane & 16) ? 8 : 0;   // C/D layout: m = vgpr + (lane>=16 ? 8 : 0)

#pragma unroll
  for (int nt = 0; nt < 8; ++nt) {
    v8f acc = {0.f, 0.f, 0.f, 0.f, 0.f, 0.f, 0.f, 0.f};
#pragma unroll
    for (int kt = 0; kt < 4; ++kt) {
      const v16bf bp = *(const v16bf*)&sP[((nt * 4 + kt) * 32 + lane) * 16];
      acc = __builtin_amdgcn_wmma_f32_16x16x32_bf16(false, a0[kt], false, bp,
                                                    (short)0, acc, false, false);
      const v16bf bm = *(const v16bf*)&sM[((nt * 4 + kt) * 32 + lane) * 16];
      acc = __builtin_amdgcn_wmma_f32_16x16x32_bf16(false, a1[kt], false, bm,
                                                    (short)0, acc, false, false);
    }
    const int nn = nt * 16 + nlo;
    const float bb = bias[nn];
#pragma unroll
    for (int v = 0; v < 8; ++v) {
      out[(size_t)(rowbase + mbase + v) * kF + nn] = acc[v] + bb;
    }
  }
}

// Kernel 2: reverse-time sigmoid scan, in place on d_out.
// out[b,t] = sigmoid(SLOPE*(wq*out[b,t+1] + combined[b,t]))  (bias already folded in)
// Thread = (b,f); chain per step kept to fma -> exp2 -> add -> rcp.
__global__ __launch_bounds__(128)
void ltl_scan_kernel(const float* __restrict__ w_qual,
                     const float* __restrict__ init_run,
                     float* __restrict__ out) {
  const int b = blockIdx.x;
  const int f = threadIdx.x;

  const float wqr = w_qual[f];
  const float wq  = fmaxf(kAlpha * wqr, wqr);          // relu_scheduled(w_qual)
  const float kw  = -kSlope * kLog2e * wq;             // folded chain coefficient
  const float kcs = -kSlope * kLog2e;                  // folds into c off the chain

  // run0 = sigmoid(SLOPE * init_run[f])
  float run = __builtin_amdgcn_rcpf(
      1.0f + __builtin_amdgcn_exp2f(-kSlope * kLog2e * init_run[f]));

  float* o = out + (size_t)b * kT * kF;

  float kc[16];
#pragma unroll
  for (int j = 0; j < 16; ++j)
    kc[j] = kcs * o[(size_t)(kT - 16 + j) * kF + f];

  for (int tb = kT - 16; tb >= 0; tb -= 16) {
    // Prefetch next chunk (independent of the recurrence chain)
    float kcn[16];
    if (tb >= 16) {
#pragma unroll
      for (int j = 0; j < 16; ++j)
        kcn[j] = kcs * o[(size_t)(tb - 16 + j) * kF + f];
    }
    // Dependent chain, t descending within the chunk
#pragma unroll
    for (int j = 15; j >= 0; --j) {
      const float t = __builtin_fmaf(kw, run, kc[j]);  // -S*log2e*(wq*run + c)
      run = __builtin_amdgcn_rcpf(1.0f + __builtin_amdgcn_exp2f(t));
      o[(size_t)(tb + j) * kF + f] = run;
    }
    if (tb >= 16) {
#pragma unroll
      for (int j = 0; j < 16; ++j) kc[j] = kcn[j];
    }
  }
}

extern "C" void kernel_launch(void* const* d_in, const int* in_sizes, int n_in,
                              void* d_out, int out_size, void* d_ws, size_t ws_size,
                              hipStream_t stream) {
  const float* traces      = (const float*)d_in[0];
  const float* w_prop      = (const float*)d_in[1];
  const float* w_metric    = (const float*)d_in[2];
  const float* init_metric = (const float*)d_in[3];
  const float* w_qual      = (const float*)d_in[4];
  const float* bias        = (const float*)d_in[5];
  const float* init_run    = (const float*)d_in[6];
  float* out = (float*)d_out;

  // Pass 1: WMMA GEMMs -> combined + bias, staged in d_out (2048 blocks x 8 waves)
  ltl_gemm_kernel<<<dim3((kB * kT) / 128), dim3(256), 0, stream>>>(
      traces, w_prop, w_metric, init_metric, bias, out);

  // Pass 2: reverse-time sigmoid scan, in place (64 blocks x 128 threads)
  ltl_scan_kernel<<<dim3(kB), dim3(kF), 0, stream>>>(w_qual, init_run, out);
}
